// Mamba_Encoder_33844342293075
// MI455X (gfx1250) — matrix operand order repaired and checked by reading, not compiled, not run on hardware
//
#include <hip/hip_runtime.h>
#include <hip/hip_bf16.h>
#include <math.h>

// Problem dims (compile-time constants from the reference)
#define B_    16
#define L_    2048
#define D_    256
#define ED_   512
#define N_    16
#define R_    16
#define KC_   4          // conv kernel width
#define NL_   3
#define NENC_ 128
#define BL_   (B_ * L_)      // 32768
#define DBC_  (R_ + 2 * N_)  // 48

typedef __attribute__((ext_vector_type(16))) __bf16 v16bf;
typedef __attribute__((ext_vector_type(8)))  __bf16 v8bf;
typedef __attribute__((ext_vector_type(8)))  float  v8f;

// ---------- small device helpers ----------
__device__ __forceinline__ float sigmoid_(float x) { return 1.0f / (1.0f + __expf(-x)); }
__device__ __forceinline__ float silu_(float x)    { return x * sigmoid_(x); }
__device__ __forceinline__ float softplus_(float x){ return (x > 20.0f) ? x : log1pf(__expf(x)); }

// float -> bf16 with round-to-nearest-even
__device__ __forceinline__ __bf16 f2bf(float f) {
    union { float f; unsigned u; } v; v.f = f;
    unsigned r = v.u + 0x7FFFu + ((v.u >> 16) & 1u);
    unsigned short hs = (unsigned short)(r >> 16);
    union { unsigned short s; __bf16 b; } o; o.s = hs;
    return o.b;
}

// ---------- generic f32 -> bf16 conversion (for weights, tiny) ----------
__global__ void cvt_bf16_kernel(const float* __restrict__ in, __bf16* __restrict__ out, int n) {
    int i = blockIdx.x * blockDim.x + threadIdx.x;
    if (i < n) out[i] = f2bf(in[i]);
}

// ---------- embedding: x = tokens @ emb_w^T + emb_b (DIN=2, pure VALU) ----------
__global__ void embed_kernel(const float* __restrict__ tokens,
                             const float* __restrict__ emb_w,
                             const float* __restrict__ emb_b,
                             float* __restrict__ x) {
    int i = blockIdx.x * blockDim.x + threadIdx.x;       // over BL*D
    int d = i % D_;
    int bl = i / D_;
    float t0 = tokens[bl * 2 + 0];
    float t1 = tokens[bl * 2 + 1];
    x[i] = fmaf(t0, emb_w[d * 2 + 0], fmaf(t1, emb_w[d * 2 + 1], emb_b[d]));
}

// ---------- RMSNorm over D=256, one row per 256-thread block; bf16 output ----------
__global__ void rmsnorm_kernel(const float* __restrict__ x,
                               const float* __restrict__ w,
                               __bf16* __restrict__ out) {
    __shared__ float red[256];
    int row = blockIdx.x;
    int t = threadIdx.x;
    float v = x[(size_t)row * D_ + t];
    red[t] = v * v;
    __syncthreads();
    #pragma unroll
    for (int s = 128; s > 0; s >>= 1) {
        if (t < s) red[t] += red[t + s];
        __syncthreads();
    }
    float inv = rsqrtf(red[0] * (1.0f / D_) + 1e-5f);
    out[(size_t)row * D_ + t] = f2bf(v * inv * w[t]);
}

// ---------- WMMA GEMM: C[M,N] = A[M,K] @ W[N,K]^T, bf16 operands, f32 accum ----------
// One wave computes a 16 x (16*NTILES) output tile: one A fragment reused
// against NTILES B fragments. Software-pipelined: fragments for step kb+32 are
// loaded before the WMMAs of step kb consume their fragments, so loads overlap
// the v_wmma chain instead of serializing behind s_wait_loadcnt 0.
template<int K, int NTILES, bool ACCUM>
__global__ __launch_bounds__(32)
void gemm_wmma_bf16(const __bf16* __restrict__ A,
                    const __bf16* __restrict__ W,
                    float* __restrict__ C, int N) {
    const int m0   = blockIdx.x << 4;
    const int n0   = blockIdx.y * (16 * NTILES);
    const int lane = threadIdx.x & 31;
    const int half = lane >> 4;                          // 0 or 1
    const int mn   = lane & 15;

    const __bf16* Arow = A + (size_t)(m0 + mn) * K;
    const __bf16* Wrow[NTILES];
    #pragma unroll
    for (int t = 0; t < NTILES; t++)
        Wrow[t] = W + (size_t)(n0 + 16 * t + mn) * K;

    v8f acc[NTILES];
    #pragma unroll
    for (int t = 0; t < NTILES; t++) acc[t] = (v8f){};

    // prologue: fragments for kb = 0
    // A fragment: lane (half, m) holds K = kb+half*16+{0..15}
    v8bf alo = *(const v8bf*)(Arow + half * 16);
    v8bf ahi = *(const v8bf*)(Arow + half * 16 + 8);
    v16bf bv[NTILES];
    #pragma unroll
    for (int t = 0; t < NTILES; t++)
        bv[t] = *(const v16bf*)(Wrow[t] + half * 16);   // K = 16*half+{0..15}

    #pragma unroll
    for (int kb = 0; kb < K; kb += 32) {
        v16bf av = __builtin_shufflevector(alo, ahi,
            0, 1, 2, 3, 4, 5, 6, 7, 8, 9, 10, 11, 12, 13, 14, 15);
        v16bf bcur[NTILES];
        #pragma unroll
        for (int t = 0; t < NTILES; t++) bcur[t] = bv[t];

        // prefetch next step's fragments (resolved at compile time by unroll)
        if (kb + 32 < K) {
            alo = *(const v8bf*)(Arow + (kb + 32) + half * 16);
            ahi = *(const v8bf*)(Arow + (kb + 40) + half * 16);
            #pragma unroll
            for (int t = 0; t < NTILES; t++)
                bv[t] = *(const v16bf*)(Wrow[t] + (kb + 32) + half * 16);
        }

        #pragma unroll
        for (int t = 0; t < NTILES; t++)
            acc[t] = __builtin_amdgcn_wmma_f32_16x16x32_bf16(
                false, av, false, bcur[t], (short)0, acc[t], false, false);
    }

    // C/D layout: element r -> row m0 + 8*half + r, col n0 + 16*t + mn
    #pragma unroll
    for (int t = 0; t < NTILES; t++) {
        float* Ct = C + (size_t)(m0 + half * 8) * N + n0 + 16 * t + mn;
        #pragma unroll
        for (int r = 0; r < 8; r++) {
            float v = acc[t][r];
            if (ACCUM) v += Ct[(size_t)r * N];
            Ct[(size_t)r * N] = v;
        }
    }
}

// ---------- causal depthwise conv (K=4) + bias + SiLU; writes f32 + bf16 ----------
__global__ void conv_silu_kernel(const float* __restrict__ xz,
                                 const float* __restrict__ w,
                                 const float* __restrict__ b,
                                 float* __restrict__ out_f32,
                                 __bf16* __restrict__ out_bf) {
    int i = blockIdx.x * blockDim.x + threadIdx.x;       // over BL*ED
    int e  = i % ED_;
    int bl = i / ED_;
    int l  = bl % L_;
    const float* wr = w + e * KC_;
    float acc = b[e];
    #pragma unroll
    for (int k = 0; k < KC_; k++) {
        int dl = l - (KC_ - 1) + k;
        if (dl >= 0)
            acc = fmaf(wr[k], xz[(size_t)(bl - (KC_ - 1) + k) * (2 * ED_) + e], acc);
    }
    float s = silu_(acc);
    out_f32[i] = s;
    out_bf[i]  = f2bf(s);
}

// ---------- delta = softplus(dt @ dt_proj_w^T + dt_proj_b)  (K=16, VALU) ----------
__global__ void dtproj_kernel(const float* __restrict__ dbc,
                              const float* __restrict__ dtw,
                              const float* __restrict__ dtb,
                              float* __restrict__ delta) {
    int i = blockIdx.x * blockDim.x + threadIdx.x;       // over BL*ED
    int e  = i % ED_;
    int bl = i / ED_;
    const float* dr = dbc + (size_t)bl * DBC_;
    const float* wr = dtw + e * R_;
    float acc = dtb[e];
    #pragma unroll
    for (int r = 0; r < R_; r++) acc = fmaf(dr[r], wr[r], acc);
    delta[i] = softplus_(acc);
}

// ---------- selective scan: block = one batch, thread = one channel e ----------
// h[16] in registers; B/C rows (shared by all channels) staged via LDS per step.
// y overwrites delta in place.
__global__ void scan_kernel(const float* __restrict__ xc,
                            float* __restrict__ delta_y,
                            const float* __restrict__ dbc,
                            const float* __restrict__ A_log,
                            const float* __restrict__ Dp) {
    __shared__ float sBC[2 * N_];
    int b = blockIdx.x;
    int e = threadIdx.x;                                 // 0..511
    float Ae[N_];
    #pragma unroll
    for (int n = 0; n < N_; n++) Ae[n] = -__expf(A_log[e * N_ + n]);
    float h[N_];
    #pragma unroll
    for (int n = 0; n < N_; n++) h[n] = 0.0f;
    float Dv = Dp[e];

    size_t base = (size_t)b * L_;
    for (int l = 0; l < L_; l++) {
        size_t bl = base + l;
        if (e < 2 * N_) sBC[e] = dbc[bl * DBC_ + R_ + e];
        __syncthreads();
        float d  = delta_y[bl * ED_ + e];
        float xv = xc[bl * ED_ + e];
        float dx = d * xv;
        float y = 0.0f;
        #pragma unroll
        for (int n = 0; n < N_; n++) {
            h[n] = fmaf(__expf(d * Ae[n]), h[n], dx * sBC[n]);
            y = fmaf(h[n], sBC[N_ + n], y);
        }
        delta_y[bl * ED_ + e] = fmaf(Dv, xv, y);
        __syncthreads();
    }
}

// ---------- y_bf = bf16( y * silu(z) ); z is second half of xz ----------
__global__ void gate_kernel(const float* __restrict__ y, const float* __restrict__ xz,
                            __bf16* __restrict__ y_bf) {
    int i = blockIdx.x * blockDim.x + threadIdx.x;       // over BL*ED
    int e  = i % ED_;
    size_t bl = (size_t)(i / ED_);
    float z = xz[bl * (2 * ED_) + ED_ + e];
    y_bf[i] = f2bf(y[i] * silu_(z));
}

// ---------- host-side orchestration ----------
extern "C" void kernel_launch(void* const* d_in, const int* in_sizes, int n_in,
                              void* d_out, int out_size, void* d_ws, size_t ws_size,
                              hipStream_t stream) {
    const float* tokens   = (const float*)d_in[0];
    const float* emb_w    = (const float*)d_in[1];
    const float* emb_b    = (const float*)d_in[2];
    const float* in_w     = (const float*)d_in[3];   // (NL, 2*ED, D)
    const float* conv_w   = (const float*)d_in[4];   // (NL, ED, 1, K)
    const float* conv_b   = (const float*)d_in[5];   // (NL, ED)
    const float* xproj_w  = (const float*)d_in[6];   // (NL, 48, ED)
    const float* dt_w     = (const float*)d_in[7];   // (NL, ED, R)
    const float* dt_b     = (const float*)d_in[8];   // (NL, ED)
    const float* A_log    = (const float*)d_in[9];   // (NL, ED, N)
    const float* D_param  = (const float*)d_in[10];  // (NL, ED)
    const float* out_w    = (const float*)d_in[11];  // (NL, D, ED)
    const float* ln_w     = (const float*)d_in[12];  // (NL, D)
    const float* normf_w  = (const float*)d_in[13];  // (D,)
    const float* head_w   = (const float*)d_in[14];  // (NENC, D)
    float* out = (float*)d_out;

    // ---- workspace layout: f32 region then bf16 region ----
    float* x     = (float*)d_ws;                       // BL*D
    float* xz    = x     + (size_t)BL_ * D_;           // BL*2*ED
    float* xc    = xz    + (size_t)BL_ * 2 * ED_;      // BL*ED
    float* dbc   = xc    + (size_t)BL_ * ED_;          // BL*48
    float* delta = dbc   + (size_t)BL_ * DBC_;         // BL*ED (reused as y)

    __bf16* xin_bf  = (__bf16*)(delta + (size_t)BL_ * ED_);  // BL*D
    __bf16* xc_bf   = xin_bf + (size_t)BL_ * D_;             // BL*ED
    __bf16* y_bf    = xc_bf  + (size_t)BL_ * ED_;            // BL*ED
    __bf16* inw_bf  = y_bf   + (size_t)BL_ * ED_;            // 2*ED*D
    __bf16* xpw_bf  = inw_bf + (size_t)2 * ED_ * D_;         // DBC*ED
    __bf16* outw_bf = xpw_bf + (size_t)DBC_ * ED_;           // D*ED
    __bf16* head_bf = outw_bf + (size_t)D_ * ED_;            // NENC*D

    const int T = 256;

    // head weights -> bf16 (once per call)
    cvt_bf16_kernel<<<(NENC_ * D_ + T - 1) / T, T, 0, stream>>>(head_w, head_bf, NENC_ * D_);

    // embedding
    embed_kernel<<<(BL_ * D_) / T, T, 0, stream>>>(tokens, emb_w, emb_b, x);

    for (int l = 0; l < NL_; l++) {
        // per-layer weights -> bf16 (tiny, L2-resident afterwards)
        cvt_bf16_kernel<<<(2 * ED_ * D_ + T - 1) / T, T, 0, stream>>>(
            in_w + (size_t)l * 2 * ED_ * D_, inw_bf, 2 * ED_ * D_);
        cvt_bf16_kernel<<<(DBC_ * ED_ + T - 1) / T, T, 0, stream>>>(
            xproj_w + (size_t)l * DBC_ * ED_, xpw_bf, DBC_ * ED_);
        cvt_bf16_kernel<<<(D_ * ED_ + T - 1) / T, T, 0, stream>>>(
            out_w + (size_t)l * D_ * ED_, outw_bf, D_ * ED_);

        // xin_bf = rmsnorm(x, ln_w[l])
        rmsnorm_kernel<<<BL_, 256, 0, stream>>>(x, ln_w + (size_t)l * D_, xin_bf);

        // xz = xin @ in_proj_w[l]^T   (32768 x 1024, K=256), 16x64 tiles
        gemm_wmma_bf16<D_, 4, false><<<dim3(BL_ / 16, (2 * ED_) / 64), 32, 0, stream>>>(
            xin_bf, inw_bf, xz, 2 * ED_);

        // xc = silu(causal depthwise conv(xz[:, :ED]))  -> f32 + bf16
        conv_silu_kernel<<<(BL_ * ED_) / T, T, 0, stream>>>(
            xz, conv_w + (size_t)l * ED_ * KC_, conv_b + (size_t)l * ED_, xc, xc_bf);

        // dbc = xc @ x_proj_w[l]^T   (32768 x 48, K=512), three 16-wide tiles
        gemm_wmma_bf16<ED_, 1, false><<<dim3(BL_ / 16, DBC_ / 16), 32, 0, stream>>>(
            xc_bf, xpw_bf, dbc, DBC_);

        // delta = softplus(dt @ dt_proj_w^T + b)
        dtproj_kernel<<<(BL_ * ED_) / T, T, 0, stream>>>(
            dbc, dt_w + (size_t)l * ED_ * R_, dt_b + (size_t)l * ED_, delta);

        // selective scan (y written in place over delta)
        scan_kernel<<<B_, ED_, 0, stream>>>(
            xc, delta, dbc, A_log + (size_t)l * ED_ * N_, D_param + (size_t)l * ED_);

        // y_bf = bf16(y * silu(z))
        gate_kernel<<<(BL_ * ED_) / T, T, 0, stream>>>(delta, xz, y_bf);

        // x += y @ out_proj_w[l]^T   (32768 x 256, K=512, accumulate), 16x64 tiles
        gemm_wmma_bf16<ED_, 4, true><<<dim3(BL_ / 16, D_ / 64), 32, 0, stream>>>(
            y_bf, outw_bf, x, D_);
    }

    // final norm + head
    rmsnorm_kernel<<<BL_, 256, 0, stream>>>(x, normf_w, xin_bf);
    gemm_wmma_bf16<D_, 4, false><<<dim3(BL_ / 16, NENC_ / 64), 32, 0, stream>>>(
        xin_bf, head_bf, out, NENC_);
}
